// Encoder_85074712199861
// MI455X (gfx1250) — matrix operand order repaired and checked by reading, not compiled, not run on hardware
//
#include <hip/hip_runtime.h>
#include <hip/hip_bf16.h>

// ---------------------------------------------------------------------------
// GNN TransformerConv x2 for MI455X (gfx1250, wave32, WMMA).
//  - Fused Q/K/V/skip projection: one A-fragment stream, 4 WMMA chains
//    (v_wmma_f32_16x16x32_f16, f32 accumulate). X read once per layer.
//  - Edge attention: 3 streaming passes, L2-resident weights, atomics.
// ---------------------------------------------------------------------------

typedef __attribute__((ext_vector_type(16))) _Float16 v16h;
typedef __attribute__((ext_vector_type(8)))  float    v8f;

#define HD   128   // hidden width (= heads*d for both layers)
#define KDIM 128   // GEMM reduction dim (node feature width, both layers)
#define EDIM 16    // edge feature dim

// ---- WMMA fragment loaders (CDNA5 16-bit layouts, ISA 7.12.2) --------------
// A (16x32, row-major in memory): lanes 0-15 hold rows M=0..15 with K 0..7 in
// elems 0..7 and K 16..23 in elems 8..15; lanes 16-31 hold K 8..15 / 24..31.
__device__ inline v16h frag_a(const float* __restrict__ X, int row0, int k0, int lane) {
  const int half = lane >> 4;
  const float* p = X + (size_t)(row0 + (lane & 15)) * KDIM + k0 + half * 8;
  v16h a;
#pragma unroll
  for (int g = 0; g < 2; ++g)
#pragma unroll
    for (int j = 0; j < 8; ++j)
      a[g * 8 + j] = (_Float16)p[g * 16 + j];
  return a;
}

// B (32x16, K x N row-major in memory): lane = output column (N = lane&15);
// lanes 0-15 carry K 0..7 / 16..23, lanes 16-31 carry K 8..15 / 24..31 (same as A).
__device__ inline v16h frag_b(const float* __restrict__ W, int k0, int col0, int lane) {
  const int c  = col0 + (lane & 15);
  const int kb = k0 + (lane >> 4) * 8;
  v16h b;
#pragma unroll
  for (int j = 0; j < 16; ++j)
    b[j] = (_Float16)W[(size_t)(kb + j + (j >> 3) * 8) * HD + c];
  return b;
}

// Fused: {Yq,Yk,Yv,Ys}[M,128] = X[M,128] @ {Wq,Wk,Wv,Ws} + {bq,bk,bv,bs}
// grid.x = M/16, block = 256 (8 waves, one 16-col tile each).
__global__ void __launch_bounds__(256)
gemm_proj4(const float* __restrict__ X,
           const float* __restrict__ Wq, const float* __restrict__ bq, float* __restrict__ Yq,
           const float* __restrict__ Wk, const float* __restrict__ bk, float* __restrict__ Yk,
           const float* __restrict__ Wv, const float* __restrict__ bv, float* __restrict__ Yv,
           const float* __restrict__ Ws, const float* __restrict__ bs, float* __restrict__ Ys) {
  const int lane = threadIdx.x & 31;
  const int wave = threadIdx.x >> 5;          // 0..7 -> 16-col tile
  const int row0 = blockIdx.x * 16;
  const int col0 = wave * 16;
  v8f aq = {}, ak = {}, av = {}, as = {};
#pragma unroll
  for (int k0 = 0; k0 < KDIM; k0 += 32) {
    if (k0 + 32 < KDIM) {  // prefetch next k-step of the A stream (global_prefetch_b8)
      const float* np = X + (size_t)(row0 + (lane & 15)) * KDIM + (k0 + 32) + (lane >> 4) * 8;
      __builtin_prefetch(np, 0, 3);
    }
    v16h a = frag_a(X, row0, k0, lane);       // loaded ONCE, feeds 4 WMMAs
    v16h b0 = frag_b(Wq, k0, col0, lane);
    aq = __builtin_amdgcn_wmma_f32_16x16x32_f16(false, a, false, b0, (short)0, aq, false, false);
    v16h b1 = frag_b(Wk, k0, col0, lane);
    ak = __builtin_amdgcn_wmma_f32_16x16x32_f16(false, a, false, b1, (short)0, ak, false, false);
    v16h b2 = frag_b(Wv, k0, col0, lane);
    av = __builtin_amdgcn_wmma_f32_16x16x32_f16(false, a, false, b2, (short)0, av, false, false);
    v16h b3 = frag_b(Ws, k0, col0, lane);
    as = __builtin_amdgcn_wmma_f32_16x16x32_f16(false, a, false, b3, (short)0, as, false, false);
  }
  // C/D layout: VGPR i -> row (lane>=16 ? 8 : 0)+i, col = lane&15
  const int c = col0 + (lane & 15);
  const int r = row0 + (lane >> 4) * 8;
  const float bvq = bq[c], bvk = bk[c], bvv = bv[c], bvs = bs[c];
#pragma unroll
  for (int i = 0; i < 8; ++i) {
    const size_t o = (size_t)(r + i) * HD + c;
    Yq[o] = aq[i] + bvq;
    Yk[o] = ak[i] + bvk;
    Yv[o] = av[i] + bvv;
    Ys[o] = as[i] + bvs;
  }
}

// ---- segment softmax helpers ----------------------------------------------
__device__ inline void atomicMaxF(float* addr, float v) {
  // monotone bit trick: signed max for >=0, unsigned min for <0. init = -inf.
  if (v >= 0.f) atomicMax((int*)addr, __float_as_int(v));
  else          atomicMin((unsigned int*)addr, (unsigned int)__float_as_int(v));
}

__global__ void init_ms(float* __restrict__ m, float* __restrict__ s, int n) {
  int i = blockIdx.x * blockDim.x + threadIdx.x;
  if (i < n) { m[i] = -__builtin_inff(); s[i] = 0.f; }
}

// Pass A: alpha[e,h] = (q[dst] . (k[src] + ea@We)) / sqrt(d); segment max.
template <int HEADS, int D>
__global__ void edge_alpha(const long long* __restrict__ ei, const float* __restrict__ ea,
                           const float* __restrict__ q, const float* __restrict__ k,
                           const float* __restrict__ We, float* __restrict__ alpha,
                           float* __restrict__ m, int E, float inv_sqrt_d) {
  int idx = blockIdx.x * blockDim.x + threadIdx.x;
  if (idx >= E * HEADS) return;
  const int e = idx / HEADS, h = idx % HEADS;
  const int src = (int)ei[e], dst = (int)ei[E + e];
  float ael[EDIM];
#pragma unroll
  for (int j = 0; j < EDIM; ++j) ael[j] = ea[(size_t)e * EDIM + j];
  const float* qi = q + (size_t)dst * HD + h * D;
  const float* ks = k + (size_t)src * HD + h * D;
  const float* Wh = We + h * D;                 // We is [EDIM, HD], L2-resident
  float acc = 0.f;
  for (int dd = 0; dd < D; ++dd) {
    float ev = 0.f;
#pragma unroll
    for (int j = 0; j < EDIM; ++j) ev += ael[j] * Wh[(size_t)j * HD + dd];
    acc += qi[dd] * (ks[dd] + ev);
  }
  acc *= inv_sqrt_d;
  alpha[idx] = acc;
  atomicMaxF(&m[(size_t)dst * HEADS + h], acc);
}

// Pass B: ex = exp(alpha - m[dst]); segment sum.
template <int HEADS>
__global__ void edge_exp(const long long* __restrict__ ei, float* __restrict__ alpha,
                         const float* __restrict__ m, float* __restrict__ s, int E) {
  int idx = blockIdx.x * blockDim.x + threadIdx.x;
  if (idx >= E * HEADS) return;
  const int e = idx / HEADS, h = idx % HEADS;
  const int dst = (int)ei[E + e];
  float ex = __expf(alpha[idx] - m[(size_t)dst * HEADS + h]);
  alpha[idx] = ex;
  atomicAdd(&s[(size_t)dst * HEADS + h], ex);
}

// Pass C: out[dst] += (ex / s[dst]) * (v[src] + ea@We)   (out pre-init by skip)
template <int HEADS, int D>
__global__ void edge_accum(const long long* __restrict__ ei, const float* __restrict__ ea,
                           const float* __restrict__ v, const float* __restrict__ We,
                           const float* __restrict__ alpha, const float* __restrict__ s,
                           float* __restrict__ out, int E) {
  int idx = blockIdx.x * blockDim.x + threadIdx.x;
  if (idx >= E * HEADS) return;
  const int e = idx / HEADS, h = idx % HEADS;
  const int src = (int)ei[e], dst = (int)ei[E + e];
  const float w = alpha[idx] / (s[(size_t)dst * HEADS + h] + 1e-16f);
  float ael[EDIM];
#pragma unroll
  for (int j = 0; j < EDIM; ++j) ael[j] = ea[(size_t)e * EDIM + j];
  const float* vs = v + (size_t)src * HD + h * D;
  const float* Wh = We + h * D;
  float* od = out + (size_t)dst * HD + h * D;
  for (int dd = 0; dd < D; ++dd) {
    float ev = 0.f;
#pragma unroll
    for (int j = 0; j < EDIM; ++j) ev += ael[j] * Wh[(size_t)j * HD + dd];
    atomicAdd(&od[dd], w * (vs[dd] + ev));
  }
}

__global__ void leaky_relu(float* __restrict__ x, int n) {
  int i = blockIdx.x * blockDim.x + threadIdx.x;
  if (i < n) { float v = x[i]; x[i] = v > 0.f ? v : 0.01f * v; }
}

// ---------------------------------------------------------------------------
extern "C" void kernel_launch(void* const* d_in, const int* in_sizes, int n_in,
                              void* d_out, int out_size, void* d_ws, size_t ws_size,
                              hipStream_t stream) {
  const float*     x   = (const float*)d_in[0];
  const long long* ei  = (const long long*)d_in[1];   // int64 [2,E]
  const float*     ea  = (const float*)d_in[2];
  const float* Wq1=(const float*)d_in[3],  *bq1=(const float*)d_in[4];
  const float* Wk1=(const float*)d_in[5],  *bk1=(const float*)d_in[6];
  const float* Wv1=(const float*)d_in[7],  *bv1=(const float*)d_in[8];
  const float* We1=(const float*)d_in[9];
  const float* Ws1=(const float*)d_in[10], *bs1=(const float*)d_in[11];
  const float* Wq2=(const float*)d_in[12], *bq2=(const float*)d_in[13];
  const float* Wk2=(const float*)d_in[14], *bk2=(const float*)d_in[15];
  const float* Wv2=(const float*)d_in[16], *bv2=(const float*)d_in[17];
  const float* We2=(const float*)d_in[18];
  const float* Ws2=(const float*)d_in[19], *bs2=(const float*)d_in[20];

  const int N = in_sizes[0] / HD;     // 50000 (multiple of 16)
  const int E = in_sizes[2] / EDIM;   // 800000
  float* out = (float*)d_out;

  // workspace layout (floats): q,k,v,h [N*128 each]; alpha [E*8]; m,s [N*8]
  float* ws    = (float*)d_ws;
  const size_t NF = (size_t)N * HD;
  float* q     = ws;
  float* kk    = q  + NF;
  float* vv    = kk + NF;
  float* hb    = vv + NF;
  float* alpha = hb + NF;
  float* mbuf  = alpha + (size_t)E * 8;
  float* sbuf  = mbuf  + (size_t)N * 8;

  const dim3 gB(N / 16), tB(256);
  const int nm = N * 8;
  const int t1 = E * 8;        // layer-1 (edge,head) threads
  const int t2 = E;            // layer-2

  // ---------------- layer 1: heads=8, d=16, concat ----------------
  gemm_proj4<<<gB, tB, 0, stream>>>(x,
      Wq1, bq1, q, Wk1, bk1, kk, Wv1, bv1, vv, Ws1, bs1, hb);  // hb = skip pre-init
  init_ms<<<(nm + 255) / 256, 256, 0, stream>>>(mbuf, sbuf, nm);
  edge_alpha<8, 16><<<(t1 + 255) / 256, 256, 0, stream>>>(ei, ea, q, kk, We1, alpha, mbuf, E, 0.25f);
  edge_exp<8><<<(t1 + 255) / 256, 256, 0, stream>>>(ei, alpha, mbuf, sbuf, E);
  edge_accum<8, 16><<<(t1 + 255) / 256, 256, 0, stream>>>(ei, ea, vv, We1, alpha, sbuf, hb, E);
  leaky_relu<<<(int)((NF + 255) / 256), 256, 0, stream>>>(hb, (int)NF);

  // ---------------- layer 2: heads=1, d=128 (mean over 1 head = identity) --
  gemm_proj4<<<gB, tB, 0, stream>>>(hb,
      Wq2, bq2, q, Wk2, bk2, kk, Wv2, bv2, vv, Ws2, bs2, out); // out = skip pre-init
  init_ms<<<(nm + 255) / 256, 256, 0, stream>>>(mbuf, sbuf, nm);
  edge_alpha<1, 128><<<(t2 + 255) / 256, 256, 0, stream>>>(ei, ea, q, kk, We2, alpha, mbuf, E, 0.08838834764831845f);
  edge_exp<1><<<(t2 + 255) / 256, 256, 0, stream>>>(ei, alpha, mbuf, sbuf, E);
  edge_accum<1, 128><<<(t2 + 255) / 256, 256, 0, stream>>>(ei, ea, vv, We2, alpha, sbuf, out, E);
}
